// SAGE2_28432683499967
// MI455X (gfx1250) — matrix operand order repaired and checked by reading, not compiled, not run on hardware
//
#include <hip/hip_runtime.h>

#define HID 160

typedef __attribute__((ext_vector_type(8)))  float  v8f;
typedef __attribute__((ext_vector_type(2)))  float  v2f;
typedef __attribute__((ext_vector_type(16))) __bf16 v16bf;
typedef __attribute__((ext_vector_type(4)))  int    v4i;

#if defined(__has_builtin)
#  if __has_builtin(__builtin_amdgcn_global_load_async_to_lds_b128) && \
      __has_builtin(__builtin_amdgcn_s_wait_asynccnt)
#    define USE_ASYNC_LDS 1
#  endif
#endif
#ifndef USE_ASYNC_LDS
#  define USE_ASYNC_LDS 0
#endif

#if USE_ASYNC_LDS
#define GLOBAL_V4I(p) ((__attribute__((address_space(1))) v4i*)(p))
#define LDS_V4I(p)    ((__attribute__((address_space(3))) v4i*)(p))
#endif

__device__ __forceinline__ float leaky(float x) { return x > 0.f ? x : 0.01f * x; }

// ---------------------------------------------------------------------------
// Encoder: H[:,colBase:colBase+32] = leaky(X @ W + b), X [n,768], W [768,32].
// 128-thread block = 4 waves = 4 row-tiles; W packed transposed (bf16) in LDS
// once per block, so B-fragments are contiguous ds loads instead of 32 strided
// global scalars per k-step. bf16 WMMA 16x16x32, fp32 accumulate.
// ---------------------------------------------------------------------------
#define ETILES 4
#define WTSTRIDE 784   // 768 + pad, keeps 32B alignment of fragment starts

__global__ __launch_bounds__(128)
void encode768_kernel(const float* __restrict__ des, const float* __restrict__ twt,
                      const float* __restrict__ xin,
                      const float* __restrict__ Wd, const float* __restrict__ bd,
                      const float* __restrict__ Wt, const float* __restrict__ bt,
                      const float* __restrict__ Wx, const float* __restrict__ bx,
                      float* __restrict__ H, int n)
{
    __shared__ __bf16 sWT[32 * WTSTRIDE];   // ~49 KB of the 320 KB WGP LDS

    const int which = blockIdx.y;
    const float* X; const float* W; const float* B; int colBase;
    if (which == 0)      { X = des; W = Wd; B = bd; colBase = 64;  }
    else if (which == 1) { X = twt; W = Wt; B = bt; colBase = 96;  }
    else                 { X = xin; W = Wx; B = bx; colBase = 128; }

    const int t = threadIdx.x;
    // pack W[768,32] -> sWT[c][k] bf16 (coalesced global reads)
    for (int i = t; i < 768 * 32; i += 128) {
        int k = i >> 5, c = i & 31;
        sWT[c * WTSTRIDE + k] = (__bf16)W[i];
    }
    __syncthreads();

    const int lane = t & 31;
    const int tile = blockIdx.x * ETILES + (t >> 5);
    const int col  = lane & 15;
    const int hi   = (lane >= 16);
    int m = tile * 16 + col;
    if (m >= n) m = n - 1;
    const float* __restrict__ row = X + (size_t)m * 768;

    v8f acc0 = {0.f,0.f,0.f,0.f,0.f,0.f,0.f,0.f};
    v8f acc1 = {0.f,0.f,0.f,0.f,0.f,0.f,0.f,0.f};

    for (int k0 = 0; k0 < 768; k0 += 32) {
        const int klo = k0 + (hi ? 16 : 0);
        __builtin_prefetch(row + klo + 256, 0, 0);   // global_prefetch_b8, streaming rows
        float4 a0 = *(const float4*)(row + klo);
        float4 a1 = *(const float4*)(row + klo + 4);
        float4 a2 = *(const float4*)(row + klo + 8);
        float4 a3 = *(const float4*)(row + klo + 12);
        v16bf a;
        a[0]=(__bf16)a0.x;  a[1]=(__bf16)a0.y;  a[2]=(__bf16)a0.z;  a[3]=(__bf16)a0.w;
        a[4]=(__bf16)a1.x;  a[5]=(__bf16)a1.y;  a[6]=(__bf16)a1.z;  a[7]=(__bf16)a1.w;
        a[8]=(__bf16)a2.x;  a[9]=(__bf16)a2.y;  a[10]=(__bf16)a2.z; a[11]=(__bf16)a2.w;
        a[12]=(__bf16)a3.x; a[13]=(__bf16)a3.y; a[14]=(__bf16)a3.z; a[15]=(__bf16)a3.w;

        const int kb = k0 + (hi ? 16 : 0);
        const __bf16* __restrict__ w0 = &sWT[col * WTSTRIDE + kb];
        const __bf16* __restrict__ w1 = &sWT[(col + 16) * WTSTRIDE + kb];
        v16bf b0, b1;
        #pragma unroll
        for (int e = 0; e < 16; ++e) { b0[e] = w0[e]; b1[e] = w1[e]; }

        acc0 = __builtin_amdgcn_wmma_f32_16x16x32_bf16(false, a, false, b0, (short)0, acc0, false, false);
        acc1 = __builtin_amdgcn_wmma_f32_16x16x32_bf16(false, a, false, b1, (short)0, acc1, false, false);
    }

    const int mrow = tile * 16 + (hi ? 8 : 0);
    const float bc0 = B[col];
    const float bc1 = B[col + 16];
    #pragma unroll
    for (int r = 0; r < 8; ++r) {
        int mm = mrow + r;
        if (mm < n) {
            H[(size_t)mm * HID + colBase + col]      = leaky(acc0[r] + bc0);
            H[(size_t)mm * HID + colBase + 16 + col] = leaky(acc1[r] + bc1);
        }
    }
}

// ---------------------------------------------------------------------------
// Small encoders: cols 0..31 = leaky(num_prop @ W_np + b_np)  (K=6)
//                 cols 32..63 = leaky(num_category @ W_nc + b_nc) (K=11)
// ---------------------------------------------------------------------------
__global__ void encode_small_kernel(const float* __restrict__ num_prop,
                                    const float* __restrict__ num_cat,
                                    const float* __restrict__ Wnp, const float* __restrict__ bnp,
                                    const float* __restrict__ Wnc, const float* __restrict__ bnc,
                                    float* __restrict__ H, int n)
{
    int idx  = blockIdx.x * blockDim.x + threadIdx.x;
    int node = idx >> 6, c = idx & 63;
    if (node >= n) return;
    if (c < 32) {
        float acc = bnp[c];
        const float* r = num_prop + (size_t)node * 6;
        #pragma unroll
        for (int k = 0; k < 6; ++k) acc += r[k] * Wnp[k * 32 + c];
        H[(size_t)node * HID + c] = leaky(acc);
    } else {
        int cc = c - 32;
        float acc = bnc[cc];
        const float* r = num_cat + (size_t)node * 11;
        #pragma unroll
        for (int k = 0; k < 11; ++k) acc += r[k] * Wnc[k * 32 + cc];
        H[(size_t)node * HID + c] = leaky(acc);
    }
}

// ---------------------------------------------------------------------------
// OUT = act( (A1 .* rowScale) @ W1 [+ A2 @ W2] + bias ), 160x160, fp32 WMMA.
// 320 threads = 10 waves; one 16-row tile per block, one 16-col block per wave.
// A-tiles staged in LDS via CDNA5 async global->LDS DMA when available;
// per-row mean scale folded into the A-fragment read (per-lane scalar).
// ---------------------------------------------------------------------------
__global__ __launch_bounds__(320)
void gemm160_kernel(const float* __restrict__ A1, const float* __restrict__ rowScale,
                    const float* __restrict__ W1,
                    const float* __restrict__ A2, const float* __restrict__ W2,
                    const float* __restrict__ bias,
                    float* __restrict__ OUT, int n, int doLeaky)
{
    __shared__ float sA[16 * HID];
    __shared__ float sB[16 * HID];
    const int tile = blockIdx.x;
    const int t    = threadIdx.x;

#if USE_ASYNC_LDS
    // 16x160 floats = 640 16B chunks; each wave-lane DMAs 16B straight to LDS.
    for (int ch = t; ch < 640; ch += 320) {
        int r = ch / 40, c4 = (ch % 40) * 4;
        int mm = tile * 16 + r; if (mm >= n) mm = n - 1;
        __builtin_amdgcn_global_load_async_to_lds_b128(
            GLOBAL_V4I(A1 + (size_t)mm * HID + c4),
            LDS_V4I(&sA[r * HID + c4]), 0, 0);
        if (A2)
            __builtin_amdgcn_global_load_async_to_lds_b128(
                GLOBAL_V4I(A2 + (size_t)mm * HID + c4),
                LDS_V4I(&sB[r * HID + c4]), 0, 0);
    }
    __builtin_amdgcn_s_wait_asynccnt(0);
    __syncthreads();
#else
    for (int i = t; i < 16 * HID; i += 320) {
        int r = i / HID, c = i % HID;
        int mm = tile * 16 + r; if (mm >= n) mm = n - 1;
        sA[i] = A1[(size_t)mm * HID + c];
        if (A2) sB[i] = A2[(size_t)mm * HID + c];
    }
    __syncthreads();
#endif

    const int lane = t & 31;
    const int c0   = (t >> 5) * 16;
    const int col  = lane & 15;
    const int koff = (lane < 16) ? 0 : 2;
    const int rowA = lane & 15;

    float rs = 1.0f;
    if (rowScale) {
        int mm = tile * 16 + rowA; if (mm >= n) mm = n - 1;
        rs = rowScale[mm];                       // 1/max(deg,1): folds mean into A1 frag
    }

    v8f acc = {0.f,0.f,0.f,0.f,0.f,0.f,0.f,0.f};
    for (int k0 = 0; k0 < HID; k0 += 4) {
        const int ka = k0 + koff;
        v2f a; a.x = sA[rowA * HID + ka] * rs;        a.y = sA[rowA * HID + ka + 1] * rs;
        v2f b; b.x = W1[(size_t)ka * HID + c0 + col]; b.y = W1[(size_t)(ka + 1) * HID + c0 + col];
        acc = __builtin_amdgcn_wmma_f32_16x16x4_f32(false, a, false, b, (short)0, acc, false, false);
    }
    if (A2) {
        for (int k0 = 0; k0 < HID; k0 += 4) {
            const int ka = k0 + koff;
            v2f a; a.x = sB[rowA * HID + ka];             a.y = sB[rowA * HID + ka + 1];
            v2f b; b.x = W2[(size_t)ka * HID + c0 + col]; b.y = W2[(size_t)(ka + 1) * HID + c0 + col];
            acc = __builtin_amdgcn_wmma_f32_16x16x4_f32(false, a, false, b, (short)0, acc, false, false);
        }
    }
    const float bc    = bias ? bias[c0 + col] : 0.f;
    const int   mbase = tile * 16 + ((lane < 16) ? 0 : 8);
    #pragma unroll
    for (int r = 0; r < 8; ++r) {
        int mm = mbase + r;
        if (mm < n) {
            float v = acc[r] + bc;
            if (doLeaky) v = leaky(v);
            OUT[(size_t)mm * HID + c0 + col] = v;
        }
    }
}

// ---------------------------------------------------------------------------
// Edge scatter: acc[dst] += H[src]; one wave per edge (32 lanes x 5 chunks).
// H (32MB) is L2-resident (192MB L2), so gather + f32 atomics stay in L2.
// ---------------------------------------------------------------------------
__global__ void scatter_kernel(const float* __restrict__ H,
                               const int* __restrict__ src, const int* __restrict__ dst,
                               float* __restrict__ acc, float* __restrict__ deg,
                               int E, int computeDeg)
{
    int t = blockIdx.x * blockDim.x + threadIdx.x;
    int e = t >> 5, lane = t & 31;
    if (e >= E) return;
    int s = src[e], d = dst[e];
    const float* hs = H + (size_t)s * HID;
    float* ad = acc + (size_t)d * HID;
    #pragma unroll
    for (int i = 0; i < 5; ++i) atomicAdd(&ad[lane + 32 * i], hs[lane + 32 * i]);
    if (computeDeg && lane == 0) atomicAdd(&deg[d], 1.0f);
}

__global__ void rdeg_kernel(float* __restrict__ deg, int n)
{
    int i = blockIdx.x * blockDim.x + threadIdx.x;
    if (i < n) deg[i] = 1.0f / fmaxf(deg[i], 1.0f);
}

__global__ void zero_kernel(float* __restrict__ p, long long n)
{
    long long i = (long long)blockIdx.x * blockDim.x + threadIdx.x;
    if (i < n) p[i] = 0.f;
}

// out[node, 0..1] = em[node] @ W_o2 + b_o2
__global__ void head_kernel(const float* __restrict__ EM, const float* __restrict__ Wo2,
                            const float* __restrict__ bo2, float* __restrict__ OUT, int n)
{
    int node = blockIdx.x * blockDim.x + threadIdx.x;
    if (node >= n) return;
    float a0 = bo2[0], a1 = bo2[1];
    const float* e = EM + (size_t)node * HID;
    for (int k = 0; k < HID; ++k) { float x = e[k]; a0 += x * Wo2[2 * k]; a1 += x * Wo2[2 * k + 1]; }
    OUT[2 * node]     = a0;
    OUT[2 * node + 1] = a1;
}

extern "C" void kernel_launch(void* const* d_in, const int* in_sizes, int n_in,
                              void* d_out, int out_size, void* d_ws, size_t ws_size,
                              hipStream_t stream)
{
    const float* x     = (const float*)d_in[0];
    const int*   ei    = (const int*)d_in[1];          // [2,E] int32
    const float* nprop = (const float*)d_in[2];
    const float* ncat  = (const float*)d_in[3];
    const float* des   = (const float*)d_in[4];
    const float* twt   = (const float*)d_in[5];
    const float* Wdes = (const float*)d_in[6];  const float* bdes = (const float*)d_in[7];
    const float* Wtw  = (const float*)d_in[8];  const float* btw  = (const float*)d_in[9];
    const float* Wtxt = (const float*)d_in[10]; const float* btxt = (const float*)d_in[11];
    const float* Wnp  = (const float*)d_in[12]; const float* bnp  = (const float*)d_in[13];
    const float* Wnc  = (const float*)d_in[14]; const float* bnc  = (const float*)d_in[15];
    const float* Win  = (const float*)d_in[16]; const float* bin  = (const float*)d_in[17];
    const float* Wl   = (const float*)d_in[18]; const float* bl   = (const float*)d_in[19];
    const float* Wr   = (const float*)d_in[20];
    const float* Wo1  = (const float*)d_in[21]; const float* bo1  = (const float*)d_in[22];
    const float* Wo2  = (const float*)d_in[23]; const float* bo2  = (const float*)d_in[24];

    const int n = in_sizes[2] / 6;     // 50000
    const int E = in_sizes[1] / 2;     // 500000
    const int* src = ei;
    const int* dst = ei + E;

    float* hA  = (float*)d_ws;                      // [n,160]
    float* hB  = hA  + (size_t)n * HID;             // [n,160]
    float* acc = hB  + (size_t)n * HID;             // [n,160] aggregation accumulator
    float* deg = acc + (size_t)n * HID;             // [n] -> becomes 1/max(deg,1)

    float* out = (float*)d_out;                     // [n,2]
    float* em  = out + (size_t)n * 2;               // [n,160]

    const int ntiles  = (n + 15) / 16;
    const int eblocks = (ntiles + ETILES - 1) / ETILES;
    const long long nh = (long long)n * HID;

    // 1. feature encoders -> hA = concat(n, c, d, tw, t) with per-feature leaky
    encode768_kernel<<<dim3(eblocks, 3), 128, 0, stream>>>(des, twt, x,
        Wdes, bdes, Wtw, btw, Wtxt, btxt, hA, n);
    encode_small_kernel<<<(n * 64 + 255) / 256, 256, 0, stream>>>(nprop, ncat,
        Wnp, bnp, Wnc, bnc, hA, n);

    // 2. hB = leaky(hA @ W_in + b_in)
    gemm160_kernel<<<ntiles, 320, 0, stream>>>(hA, nullptr, Win, nullptr, nullptr, bin, hB, n, 1);

    // 3. conv1: acc = segment_sum(hB[src] -> dst); deg; hA = mean@W_l + b_l + hB@W_r
    zero_kernel<<<(int)((nh + 255) / 256), 256, 0, stream>>>(acc, nh);
    zero_kernel<<<(n + 255) / 256, 256, 0, stream>>>(deg, n);
    scatter_kernel<<<(E * 32 + 255) / 256, 256, 0, stream>>>(hB, src, dst, acc, deg, E, 1);
    rdeg_kernel<<<(n + 255) / 256, 256, 0, stream>>>(deg, n);
    gemm160_kernel<<<ntiles, 320, 0, stream>>>(acc, deg, Wl, hB, Wr, bl, hA, n, 0);

    // 4. conv2 (same weights): hB = mean@W_l + b_l + hA@W_r
    zero_kernel<<<(int)((nh + 255) / 256), 256, 0, stream>>>(acc, nh);
    scatter_kernel<<<(E * 32 + 255) / 256, 256, 0, stream>>>(hA, src, dst, acc, deg, E, 0);
    gemm160_kernel<<<ntiles, 320, 0, stream>>>(acc, deg, Wl, hA, Wr, bl, hB, n, 0);

    // 5. em = leaky(hB @ W_o1 + b_o1)  (written straight into d_out tail)
    gemm160_kernel<<<ntiles, 320, 0, stream>>>(hB, nullptr, Wo1, nullptr, nullptr, bo1, em, n, 1);

    // 6. out = em @ W_o2 + b_o2
    head_kernel<<<(n + 255) / 256, 256, 0, stream>>>(em, Wo2, bo2, out, n);
}
